// TokenRoutedMLP_39968965656877
// MI455X (gfx1250) — matrix operand order repaired and checked by reading, not compiled, not run on hardware
//
#include <hip/hip_runtime.h>
#include <hip/hip_bf16.h>

#define HDIM 2048
#define NEXP 16
#define ILOC 512
#define TWOI 1024
#define NTOK 4096
#define VOCAB 32000

// gate_up frag grid: per expert 64 K-chunks x 64 col-tiles, 512 ushorts (1KB) per frag
#define GU_CHUNKS 64
#define GU_TILES  64
// down frag grid: per expert 16 K-chunks x 128 col-tiles
#define DN_CHUNKS 16
#define DN_TILES  128

typedef __attribute__((ext_vector_type(16))) __bf16        v16bf;
typedef __attribute__((ext_vector_type(8)))  float         v8f;
typedef __attribute__((ext_vector_type(4)))  unsigned int  v4u;
typedef __attribute__((ext_vector_type(2)))  unsigned int  v2u;

union Frag {
    v16bf v;
    v4u   q[2];
    unsigned short s[16];
};

__device__ __forceinline__ unsigned short f2bf(float f) {
    union { __bf16 b; unsigned short u; } cv;
    cv.b = (__bf16)f;          // native cvt to bf16
    return cv.u;
}

__device__ __forceinline__ v8f wmma_bf16(v16bf a, v16bf b, v8f c) {
    // (neg_a, A, neg_b, B, c_mod, C, reuse_a, reuse_b)
    return __builtin_amdgcn_wmma_f32_16x16x32_bf16(false, a, false, b, (short)0, c, false, false);
}

// Fragment stored lane-major: frag[lane][j], 32 lanes x 16 ushorts = 2x b128 per lane.
__device__ __forceinline__ v16bf load_frag(const unsigned short* p, int lane) {
    Frag f;
    const v4u* src = (const v4u*)(p + (lane << 4));
    f.q[0] = src[0];
    f.q[1] = src[1];
    return f.v;
}

__device__ __forceinline__ size_t gfrag_off(int e, int chunk, int tile) {
    return (((size_t)e * GU_CHUNKS + chunk) * GU_TILES + tile) << 9;   // *512 ushorts
}
__device__ __forceinline__ size_t dfrag_off(int e, int chunk, int tile) {
    return (((size_t)e * DN_CHUNKS + chunk) * DN_TILES + tile) << 9;
}

// A layout helper (contiguous k order, matches B frags): lane = m + 16*h, s[j]:
//   k_local in [16h, 16h+16)     -> j = k_local & 15
__device__ __forceinline__ void a_slot(int k_local, int m, int* lane_out, int* j_out) {
    int h = (k_local >> 4) & 1;
    *j_out    = k_local & 15;
    *lane_out = m + (h << 4);
}

// ---------------- weight pre-conversion: f32 -> bf16, swizzled to B-frag lane order ----
// B layout: lane L -> N = L&15, khalf = L>>4 ; s[j] holds K = 16*khalf + j.
__global__ __launch_bounds__(256) void convert_gate_up_kernel(const float* __restrict__ w,
                                                              unsigned short* __restrict__ out) {
    const size_t total = (size_t)NEXP * GU_CHUNKS * GU_TILES * 512;
    for (size_t o = (size_t)blockIdx.x * 256 + threadIdx.x; o < total;
         o += (size_t)gridDim.x * 256) {
        int j     = (int)(o & 15);
        int lane  = (int)((o >> 4) & 31);
        int tile  = (int)((o >> 9) & 63);
        int chunk = (int)((o >> 15) & 63);
        int e     = (int)(o >> 21);
        int k = chunk * 32 + (lane >> 4) * 16 + j;
        int c = tile * 16 + (lane & 15);
        out[o] = f2bf(w[((size_t)e * HDIM + k) * TWOI + c]);
    }
}

__global__ __launch_bounds__(256) void convert_down_kernel(const float* __restrict__ w,
                                                           unsigned short* __restrict__ out) {
    const size_t total = (size_t)NEXP * DN_CHUNKS * DN_TILES * 512;
    for (size_t o = (size_t)blockIdx.x * 256 + threadIdx.x; o < total;
         o += (size_t)gridDim.x * 256) {
        int j     = (int)(o & 15);
        int lane  = (int)((o >> 4) & 31);
        int tile  = (int)((o >> 9) & 127);
        int chunk = (int)((o >> 16) & 15);
        int e     = (int)(o >> 20);
        int k = chunk * 32 + (lane >> 4) * 16 + j;
        int c = tile * 16 + (lane & 15);
        out[o] = f2bf(w[((size_t)e * ILOC + k) * HDIM + c]);
    }
}

__global__ void zero_counts_kernel(int* counts) {
    if (threadIdx.x < NEXP) counts[threadIdx.x] = 0;
}

// ---------------- routing: one wave per token ----------------
__global__ __launch_bounds__(256) void route_kernel(const int* __restrict__ token_ids,
                                                    const float* __restrict__ mu,
                                                    const float* __restrict__ wr,
                                                    int* __restrict__ counts,
                                                    int* __restrict__ bucket) {
    int tid  = threadIdx.x;
    int lane = tid & 31;
    int wv   = tid >> 5;
    int n    = blockIdx.x * 8 + wv;
    if (n >= NTOK) return;

    float acc[NEXP];
#pragma unroll
    for (int e = 0; e < NEXP; ++e) acc[e] = 0.0f;

    const float* murow = mu + (size_t)n * HDIM;
    for (int h = lane; h < HDIM; h += 32) {
        float m = murow[h];
#pragma unroll
        for (int e = 0; e < NEXP; ++e) acc[e] += m * wr[e * HDIM + h];
    }
#pragma unroll
    for (int e = 0; e < NEXP; ++e) {
#pragma unroll
        for (int off = 16; off > 0; off >>= 1)
            acc[e] += __shfl_xor(acc[e], off, 32);
    }
    if (lane == 0) {
        int tk = token_ids[n];
        tk = tk < 0 ? 0 : (tk > VOCAB - 1 ? VOCAB - 1 : tk);
        acc[tk % NEXP] += 10.0f;   // BASE_ROUTING_SCALE
        int best = 0;
        float bv = acc[0];
#pragma unroll
        for (int e = 1; e < NEXP; ++e) {
            if (acc[e] > bv) { bv = acc[e]; best = e; }
        }
        int pos = atomicAdd(&counts[best], 1);
        bucket[best * NTOK + pos] = n;
    }
}

// ---------------- main MoE kernel: 16-token tile x expert, 8 waves ----------------
__global__ __launch_bounds__(256) void moe_kernel(const float* __restrict__ x,
                                                  const unsigned short* __restrict__ wgb,
                                                  const unsigned short* __restrict__ wdb,
                                                  const int* __restrict__ counts,
                                                  const int* __restrict__ bucket,
                                                  float* __restrict__ out) {
    const int e = blockIdx.y;
    const int t = blockIdx.x;
    const int cnt = counts[e];
    if (t * 16 >= cnt) return;    // uniform across block

    __shared__ int rows_s[16];
    __shared__ unsigned short xsf[2][1024];     // double-buffered x chunk, A-frag order (4KB)
    __shared__ unsigned short hs[16][32][16];   // h: 16 K-chunks in A-frag lane order (16KB)

    const int tid  = threadIdx.x;
    const int lane = tid & 31;
    const int wv   = tid >> 5;
    const int nn   = lane & 15;
    const int half = lane >> 4;

    if (tid < 16) {
        int g = t * 16 + tid;
        rows_s[tid] = (g < cnt) ? bucket[e * NTOK + g] : -1;
    }
    __syncthreads();

    // ---- loop-invariant staging setup: thread -> (row m, 4 contiguous cols) ----
    const int m_ld = tid >> 4;               // token row 0..15
    const int c_ld = (tid & 15) << 2;        // col-in-chunk 0,4,...,60
    int slot;                                // offset in one 1024-ushort buffer
    {
        int la, jj;
        a_slot(c_ld & 31, m_ld, &la, &jj);   // 4 consecutive cols share (la, jj..jj+3)
        slot = ((c_ld >> 5) << 9) + (la << 4) + jj;
    }
    int r_ld  = rows_s[m_ld];
    int rs_ld = (r_ld < 0) ? rows_s[0] : r_ld;   // rows_s[0] always valid
    const float* xrow = x + (size_t)rs_ld * HDIM + c_ld;

    // output rows for this lane's C-slots (loop-invariant)
    int orow[8];
#pragma unroll
    for (int v = 0; v < 8; ++v) orow[v] = rows_s[v + 8 * half];

    auto stage = [&](int i, int buf) {   // stage K-chunk i (64 wide) into xsf[buf]
        const float4 xv = *(const float4*)(xrow + (i << 6));
        union { unsigned short s[4]; v2u u; } pk;
        pk.s[0] = f2bf(xv.x);
        pk.s[1] = f2bf(xv.y);
        pk.s[2] = f2bf(xv.z);
        pk.s[3] = f2bf(xv.w);
        *(v2u*)(&xsf[buf][slot]) = pk.u;     // single ds_store_b64
    };

    v8f accg[4], accu[4];
#pragma unroll
    for (int q = 0; q < 4; ++q) {
        accg[q] = (v8f){0.f, 0.f, 0.f, 0.f, 0.f, 0.f, 0.f, 0.f};
        accu[q] = (v8f){0.f, 0.f, 0.f, 0.f, 0.f, 0.f, 0.f, 0.f};
    }

    // ---------------- Phase A: gate_up GEMM over K=2048, software pipelined --------
    stage(0, 0);
    for (int i = 0; i < 32; ++i) {           // 32 chunks of 64
        __syncthreads();
        if (i + 1 < 32) stage(i + 1, (i + 1) & 1);

        const int buf = i & 1;
        v16bf A0 = load_frag(&xsf[buf][0],   lane);
        v16bf A1 = load_frag(&xsf[buf][512], lane);
        const int chunkA = i << 1;

#pragma unroll
        for (int q = 0; q < 4; ++q) {
            int it = wv * 4 + q;             // intermediate tile 0..31
            const unsigned short* bg0 = wgb + gfrag_off(e, chunkA,     it);
            const unsigned short* bg1 = wgb + gfrag_off(e, chunkA + 1, it);
            const unsigned short* bu0 = wgb + gfrag_off(e, chunkA,     32 + it);
            const unsigned short* bu1 = wgb + gfrag_off(e, chunkA + 1, 32 + it);
            if (chunkA + 2 < GU_CHUNKS) {
                __builtin_prefetch(wgb + gfrag_off(e, chunkA + 2, it), 0, 1);
                __builtin_prefetch(wgb + gfrag_off(e, chunkA + 2, 32 + it), 0, 1);
            }
            accg[q] = wmma_bf16(A0, load_frag(bg0, lane), accg[q]);
            accg[q] = wmma_bf16(A1, load_frag(bg1, lane), accg[q]);
            accu[q] = wmma_bf16(A0, load_frag(bu0, lane), accu[q]);
            accu[q] = wmma_bf16(A1, load_frag(bu1, lane), accu[q]);
        }
    }

    // ---------------- SwiGLU -> hs (A-frag lane order, keyed by K index i) -------
#pragma unroll
    for (int q = 0; q < 4; ++q) {
        int it = wv * 4 + q;
#pragma unroll
        for (int v = 0; v < 8; ++v) {
            float g = accg[q][v];
            float u = accu[q][v];
            float hval = (g / (1.0f + __expf(-g))) * u;   // silu(g) * u
            int m = v + 8 * half;        // token row
            int i = it * 16 + nn;        // intermediate index = K of phase B
            int la, jj;
            a_slot(i & 31, m, &la, &jj);
            hs[i >> 5][la][jj] = f2bf(hval);
        }
    }
    __syncthreads();

    // ---------------- Phase B: down GEMM, A frags resident in registers -----------
    v16bf Afr[16];
#pragma unroll
    for (int c = 0; c < 16; ++c) Afr[c] = load_frag(&hs[c][0][0], lane);

    for (int jt = 0; jt < 16; ++jt) {
        int ct = jt * 8 + wv;            // output column tile 0..127
        v8f acc = (v8f){0.f, 0.f, 0.f, 0.f, 0.f, 0.f, 0.f, 0.f};
#pragma unroll
        for (int c = 0; c < 16; ++c) {
            v16bf B = load_frag(wdb + dfrag_off(e, c, ct), lane);
            acc = wmma_bf16(Afr[c], B, acc);
        }
#pragma unroll
        for (int v = 0; v < 8; ++v) {
            if (orow[v] >= 0) out[(size_t)orow[v] * HDIM + ct * 16 + nn] = acc[v];
        }
    }
}

extern "C" void kernel_launch(void* const* d_in, const int* in_sizes, int n_in,
                              void* d_out, int out_size, void* d_ws, size_t ws_size,
                              hipStream_t stream) {
    (void)in_sizes; (void)n_in; (void)out_size; (void)ws_size;
    const float* x         = (const float*)d_in[0];
    const int*   token_ids = (const int*)d_in[1];
    const float* mu        = (const float*)d_in[2];
    const float* gate_up   = (const float*)d_in[3];
    const float* down      = (const float*)d_in[4];
    const float* wr        = (const float*)d_in[5];
    float* out             = (float*)d_out;

    char* ws = (char*)d_ws;
    int* counts = (int*)ws;                                   // 256B reserved
    int* bucket = (int*)(ws + 256);                           // 16*4096*4 = 256KB
    unsigned short* wgb = (unsigned short*)(ws + (1 << 19));  // @512KB: 64MB bf16 frags
    unsigned short* wdb = (unsigned short*)(ws + (1 << 19) + (64u << 20)); // +64MB: 32MB

    zero_counts_kernel<<<1, 32, 0, stream>>>(counts);
    convert_gate_up_kernel<<<4096, 256, 0, stream>>>(gate_up, wgb);
    convert_down_kernel<<<2048, 256, 0, stream>>>(down, wdb);
    route_kernel<<<NTOK / 8, 256, 0, stream>>>(token_ids, mu, wr, counts, bucket);

    dim3 grid(NTOK / 16, NEXP);
    moe_kernel<<<grid, 256, 0, stream>>>(x, wgb, wdb, counts, bucket, out);
}
